// ContrastiveLoss_44238163149311
// MI455X (gfx1250) — compile-verified, matrix operand order checked by reading, not run on hardware
//
#include <hip/hip_runtime.h>
#include <hip/hip_bf16.h>
#include <hip/hip_fp16.h>

typedef __attribute__((ext_vector_type(16))) _Float16 v16h;
typedef __attribute__((ext_vector_type(8)))  _Float16 v8h;
typedef __attribute__((ext_vector_type(8)))  float    v8f;

#define B_ROWS   4096
#define DIM      256
#define NROWS    8192                 // 2B
#define INV_T    (1.0f / 0.07f)
#define TILE_K   32
#define KSTEPS   (DIM / TILE_K)       // 8
#define COLS_PER_BLOCK 512
#define NCB      (NROWS / COLS_PER_BLOCK)   // 16 col-blocks
#define LDS_STRIDE 264                // 256 halves + 8 pad (bank-conflict avoidance)
#define LDSB_COLS 32                  // columns staged per iteration (2 WMMA tiles)
#define NITER    (COLS_PER_BLOCK / LDSB_COLS)  // 16

// Scheduler hints: best-effort instruction-group interleaving (no-op if absent).
#if __has_builtin(__builtin_amdgcn_sched_group_barrier)
#define SCHED_GROUP(mask, size) __builtin_amdgcn_sched_group_barrier((mask), (size), 0)
#else
#define SCHED_GROUP(mask, size)
#endif
#define SG_WMMA   0x008   // MFMA/WMMA
#define SG_VMREAD 0x020   // VMEM read
#define SG_DSREAD 0x100   // DS read

// ---------------------------------------------------------------------------
// Block-wide (256-thread) sum reduction. Wave32: shuffle within wave, LDS across.
__device__ __forceinline__ float block_sum_256(float v, float* red) {
    #pragma unroll
    for (int m = 16; m > 0; m >>= 1) v += __shfl_xor(v, m, 32);
    const int lane = threadIdx.x & 31, wave = threadIdx.x >> 5;
    __syncthreads();                       // guard red[] against previous use
    if (lane == 0) red[wave] = v;
    __syncthreads();
    float s = red[0];
    #pragma unroll
    for (int w = 1; w < 8; ++w) s += red[w];
    return s;
}

// ---------------------------------------------------------------------------
// Kernel 1: L2-normalize rows of anchor & contrast, emit f16 feat = [f; g],
// and f32 positives d[i] = f_i . g_i (kept in full precision).
__global__ void normalize_kernel(const float* __restrict__ anchor,
                                 const float* __restrict__ contrast,
                                 _Float16* __restrict__ featH,
                                 float* __restrict__ pos) {
    __shared__ float red[8];
    const int i = blockIdx.x;        // row 0..4095
    const int d = threadIdx.x;       // 0..255 == DIM
    const float a = anchor[i * DIM + d];
    const float c = contrast[i * DIM + d];

    const float sa = block_sum_256(a * a, red);
    const float fa = a / fmaxf(sqrtf(sa), 1e-12f);
    const float sc = block_sum_256(c * c, red);
    const float fc = c / fmaxf(sqrtf(sc), 1e-12f);

    featH[i * DIM + d]            = (_Float16)fa;
    featH[(B_ROWS + i) * DIM + d] = (_Float16)fc;

    const float p = block_sum_256(fa * fc, red);
    if (d == 0) pos[i] = p;
}

// ---------------------------------------------------------------------------
// B-fragment loader: 16x32 f16 fragment layout, two 16B LDS reads per lane.
__device__ __forceinline__ v16h ldfrag(const _Float16* p, int kk) {
    union { v16h v; v8h h[2]; } u;
    u.h[0] = *(const v8h*)(p + kk * TILE_K);
    u.h[1] = *(const v8h*)(p + kk * TILE_K + 16);
    return u.v;
}

// ---------------------------------------------------------------------------
// Kernel 2: fused sim = feat @ feat^T  ->  exp(sim/T) (diag masked) -> row sums.
// Block = 8 waves x 32 = 256 threads, 128 rows x COLS_PER_BLOCK cols.
// Each wave: 16 rows, A fragments resident in VGPRs for the whole column sweep.
// Per iteration (branchless, single scheduling region):
//   - 4 global loads of next tile-pair issued first (hidden behind WMMAs)
//   - two independent K=256 WMMA chains; B frags loaded one k-step ahead
//     (sched_group_barrier pattern: [4 ds_read] before each [2 wmma])
//   - exp/diag/row-sum accumulation
//   - ds_store into the other LDS buffer, one barrier per iteration
__launch_bounds__(256)
__global__ void simclr_gemm_kernel(const _Float16* __restrict__ featH,
                                   float* __restrict__ partial) {
    __shared__ _Float16 ldsB[2][LDSB_COLS * LDS_STRIDE];   // 2 x 16896 B

    const int tid  = threadIdx.x;
    const int wave = tid >> 5;
    const int lane = tid & 31;
    const int rowBase  = blockIdx.y * 128 + wave * 16;   // this wave's 16 rows
    const int colStart = blockIdx.x * COLS_PER_BLOCK;

    // Staging coords: thread t -> column sc (0..31), 64B chunk at half-offset skc.
    const int sc  = tid >> 3;
    const int skc = (tid & 7) * 32;

    // --- Resident A fragments: 16 rows x 256 K (f16), ISA 16x32 A layout:
    // lane = 16*kGroup + M ; halves 0..7 -> K = kHalf..+7, 8..15 -> +16.
    const int arow  = rowBase + (lane & 15);
    const int kHalf = (lane >> 4) * 8;                   // 0 or 8
    const _Float16* aptr = featH + (size_t)arow * DIM;
    v16h afrag[KSTEPS];
    #pragma unroll
    for (int kk = 0; kk < KSTEPS; ++kk) {
        union { v16h v; v8h h[2]; } u;
        u.h[0] = *(const v8h*)(aptr + kk * TILE_K + kHalf);
        u.h[1] = *(const v8h*)(aptr + kk * TILE_K + kHalf + 16);
        afrag[kk] = u.v;
    }

    v8f esum = {};                                       // per-row running exp-sums
    const int mBase = rowBase + ((lane >> 4) << 3);      // C/D layout: M = mBase + r

    // Prologue: stage tile-pair 0 into buffer 0.
    {
        const _Float16* src = featH + (size_t)(colStart + sc) * DIM + skc;
        _Float16* dst = &ldsB[0][sc * LDS_STRIDE + skc];
        #pragma unroll
        for (int q = 0; q < 4; ++q) ((v8h*)dst)[q] = ((const v8h*)src)[q];
    }
    __syncthreads();

    for (int it = 0; it < NITER; ++it) {
        const int cur     = it & 1;
        const int colBase = colStart + it * LDSB_COLS;
        // Clamp so the last iteration re-stages (harmlessly) instead of branching:
        const int nextIt  = (it + 1 < NITER) ? (it + 1) : it;

        // Issue next tile-pair's global loads early (consumed after the WMMAs).
        const _Float16* gsrc =
            featH + (size_t)(colStart + nextIt * LDSB_COLS + sc) * DIM + skc;
        v8h g0 = ((const v8h*)gsrc)[0];
        v8h g1 = ((const v8h*)gsrc)[1];
        v8h g2 = ((const v8h*)gsrc)[2];
        v8h g3 = ((const v8h*)gsrc)[3];

        // Two independent K=256 WMMA chains (cols colBase..+15 and +16..+31),
        // B fragments for step kk+1 prefetched before step kk's math.
        const _Float16* bbase = &ldsB[cur][(lane & 15) * LDS_STRIDE + kHalf];
        v8f acc0 = {}, acc1 = {};
        v16h b0 = ldfrag(bbase, 0);
        v16h b1 = ldfrag(bbase + 16 * LDS_STRIDE, 0);
        #pragma unroll
        for (int kk = 0; kk < KSTEPS; ++kk) {
            v16h n0, n1;
            if (kk + 1 < KSTEPS) {
                n0 = ldfrag(bbase, kk + 1);
                n1 = ldfrag(bbase + 16 * LDS_STRIDE, kk + 1);
            }
            acc0 = __builtin_amdgcn_wmma_f32_16x16x32_f16(
                       false, afrag[kk], false, b0, (short)0, acc0, false, false);
            acc1 = __builtin_amdgcn_wmma_f32_16x16x32_f16(
                       false, afrag[kk], false, b1, (short)0, acc1, false, false);
            if (kk + 1 < KSTEPS) { b0 = n0; b1 = n1; }
        }

        // Requested schedule: stage loads first, then per k-step the next step's
        // 4 DS reads ahead of the 2 WMMAs consuming the current fragments.
        SCHED_GROUP(SG_VMREAD, 4);          // 4 global staging loads
        SCHED_GROUP(SG_DSREAD, 4);          // k=0 fragment loads
        #pragma unroll
        for (int kk = 0; kk < KSTEPS; ++kk) {
            if (kk + 1 < KSTEPS) SCHED_GROUP(SG_DSREAD, 4);  // k+1 frags ahead
            SCHED_GROUP(SG_WMMA, 2);                         // step-k WMMA pair
        }

        // exp(sim/T), mask diagonal, accumulate per-row sums (shared esum).
        const int gcol0 = colBase + (lane & 15);
        const int gcol1 = gcol0 + 16;
        #pragma unroll
        for (int r = 0; r < 8; ++r) {
            const int grow = mBase + r;
            float e0 = __expf(acc0[r] * INV_T);
            float e1 = __expf(acc1[r] * INV_T);
            if (grow == gcol0) e0 = 0.0f;
            if (grow == gcol1) e1 = 0.0f;
            esum[r] += e0 + e1;
        }

        // Store staged data into the other buffer; single barrier per iteration.
        {
            _Float16* dst = &ldsB[cur ^ 1][sc * LDS_STRIDE + skc];
            ((v8h*)dst)[0] = g0;
            ((v8h*)dst)[1] = g1;
            ((v8h*)dst)[2] = g2;
            ((v8h*)dst)[3] = g3;
        }
        __syncthreads();
    }

    // Reduce each row across the 16 lanes of its half-wave (xor masks < 16).
    #pragma unroll
    for (int r = 0; r < 8; ++r) {
        float v = esum[r];
        v += __shfl_xor(v, 1, 32);
        v += __shfl_xor(v, 2, 32);
        v += __shfl_xor(v, 4, 32);
        v += __shfl_xor(v, 8, 32);
        esum[r] = v;
    }
    if ((lane & 15) == 0) {
        float* p = partial + (size_t)blockIdx.x * NROWS + mBase;
        #pragma unroll
        for (int r = 0; r < 8; ++r) p[r] = esum[r];   // deterministic, no atomics
    }
}

// ---------------------------------------------------------------------------
// Kernel 3: denom[i] = sum of 16 col-block partials; loss = mean(log(denom) - pos/T)
__global__ void loss_reduce_kernel(const float* __restrict__ pos,
                                   const float* __restrict__ partial,
                                   float* __restrict__ out) {
    __shared__ float red[8];
    float s = 0.0f;
    for (int i = threadIdx.x; i < NROWS; i += 256) {
        float den = 0.0f;
        #pragma unroll
        for (int cb = 0; cb < NCB; ++cb) den += partial[(size_t)cb * NROWS + i];
        s += logf(den) - pos[i & (B_ROWS - 1)] * INV_T;
    }
    #pragma unroll
    for (int m = 16; m > 0; m >>= 1) s += __shfl_xor(s, m, 32);
    const int lane = threadIdx.x & 31, wave = threadIdx.x >> 5;
    if (lane == 0) red[wave] = s;
    __syncthreads();
    if (threadIdx.x == 0) {
        float t = 0.0f;
        #pragma unroll
        for (int w = 0; w < 8; ++w) t += red[w];
        out[0] = t / (float)NROWS;
    }
}

// ---------------------------------------------------------------------------
extern "C" void kernel_launch(void* const* d_in, const int* in_sizes, int n_in,
                              void* d_out, int out_size, void* d_ws, size_t ws_size,
                              hipStream_t stream) {
    const float* anchor   = (const float*)d_in[0];
    const float* contrast = (const float*)d_in[1];

    // ws layout: feat f16 [8192*256] (4 MiB) | pos f32 [4096] | partial f32 [16*8192]
    _Float16* featH   = (_Float16*)d_ws;
    float*    pos     = (float*)((char*)d_ws + (size_t)NROWS * DIM * sizeof(_Float16));
    float*    partial = pos + B_ROWS;
    float*    out     = (float*)d_out;

    normalize_kernel<<<B_ROWS, 256, 0, stream>>>(anchor, contrast, featH, pos);

    dim3 grid(NCB, NROWS / 128);   // (16, 64)
    simclr_gemm_kernel<<<grid, 256, 0, stream>>>(featH, partial);

    loss_reduce_kernel<<<1, 256, 0, stream>>>(pos, partial, out);
}
